// DKT_89034672046889
// MI455X (gfx1250) — compile-verified, matrix operand order checked by reading, not run on hardware
//
#include <hip/hip_runtime.h>
#include <hip/hip_bf16.h>
#include <stdint.h>

// ---------------- problem constants ----------------
#define B_SZ 1024
#define T_SZ 39
#define BT   (B_SZ * T_SZ)   // 39936 = 2496 * 16
#define NC   123             // NUM_C == H
#define EMBD 256
#define IN1  379             // NC + EMB
#define KP_C 128             // padded 123
#define KP_X 384             // padded 379
#define KP_H 512             // padded 502
#define NP   128             // padded output dim 123

typedef _Float16 half_t;
typedef __attribute__((ext_vector_type(8)))  _Float16 h8;
typedef __attribute__((ext_vector_type(16))) _Float16 v16h;
typedef __attribute__((ext_vector_type(8)))  float    v8f;
typedef __attribute__((ext_vector_type(4)))  unsigned int u32x4;
typedef __attribute__((ext_vector_type(8)))  int i32x8;
typedef __attribute__((ext_vector_type(4)))  int i32x4;

union F16Frag { v16h v; h8 h[2]; };

__device__ __forceinline__ float sigf(float x) { return 1.0f / (1.0f + __expf(-x)); }
__device__ __forceinline__ float tanhfast(float x) {
  float e = __expf(2.0f * x);
  return (e - 1.0f) / (e + 1.0f);
}
__device__ __forceinline__ v8f zero8() {
  v8f v;
  #pragma unroll
  for (int i = 0; i < 8; ++i) v[i] = 0.0f;
  return v;
}

// ---------------- Tensor Data Mover staging (guarded; fallback = sync copy) ---
#if defined(__has_builtin)
#if __has_builtin(__builtin_amdgcn_tensor_load_to_lds) && \
    __has_builtin(__builtin_amdgcn_s_wait_tensorcnt)
#define USE_TDM 1
#endif
#endif

#ifdef USE_TDM
// 1-row 2D descriptor: copy nbytes (multiple of 8, <=512KB) global -> LDS.
// D# layout per CDNA5 ISA ch.8: group0 {count=1, lds_addr, global_addr, type=2},
// group1 {data_size=8B, tensor_dim0=tile_dim0=n8, dims1=1, stride0=n8}.
__device__ __forceinline__ void tdm_load_lds(const void* gsrc, unsigned lds_byte_off,
                                             unsigned nbytes) {
  unsigned n8 = nbytes >> 3;  // 8-byte elements
  unsigned long long ga = (unsigned long long)gsrc;
  u32x4 g0;
  g0[0] = 1u;                                   // count=1, user descriptor
  g0[1] = lds_byte_off;                         // LDS byte address
  g0[2] = (unsigned)(ga & 0xffffffffu);         // global_addr[31:0]
  g0[3] = (unsigned)(ga >> 32) | (2u << 30);    // global_addr[56:32] | type=2
  i32x8 g1;
  g1[0] = (int)(3u << 16);                      // wg_mask=0, data_size=3 (8B)
  g1[1] = (int)((n8 & 0xffffu) << 16);          // tensor_dim0[15:0]
  g1[2] = (int)((n8 >> 16) | (1u << 16));       // tensor_dim0[31:16] | tensor_dim1=1
  g1[3] = (int)((n8 & 0xffffu) << 16);          // tensor_dim1 hi=0 | tile_dim0
  g1[4] = 1;                                    // tile_dim1=1, tile_dim2=0
  g1[5] = (int)n8;                              // tensor_dim0_stride[31:0]
  g1[6] = 0;                                    // stride0 hi | stride1 lo
  g1[7] = 0;                                    // stride1 hi
  i32x4 gz;
  gz[0] = 0; gz[1] = 0; gz[2] = 0; gz[3] = 0;
#if __clang_major__ >= 23
  i32x8 gz8;
  #pragma unroll
  for (int i = 0; i < 8; ++i) gz8[i] = 0;
  __builtin_amdgcn_tensor_load_to_lds(g0, g1, gz, gz, gz8, 0);
#else
  __builtin_amdgcn_tensor_load_to_lds(g0, g1, gz, gz, 0);
#endif
}
#endif

// Load one 16x32 f16 WMMA operand fragment (A from X rows, or B from W rows,
// since B[k][n] = W[n][k] for D = X * W^T and both stripe K the same way).
// Lane layout (CDNA5 ISA 7.12.2): lane&15 selects row, (lane>>4)*8 selects the
// K sub-block; elements 0..7 = K[koff..koff+7], 8..15 = K[16+koff..16+koff+7].
__device__ __forceinline__ v16h load_frag(const half_t* __restrict__ base, int ld,
                                          int r0, int kb, int lane) {
  int rr = r0 + (lane & 15);
  int koff = (lane >> 4) * 8;
  const half_t* p = base + (size_t)rr * ld + kb + koff;
  F16Frag f;
  f.h[0] = *(const h8*)(p);
  f.h[1] = *(const h8*)(p + 16);
  return f.v;
}

// ---------------- weight / activation staging ----------------
__global__ void k_convert_w(const float* __restrict__ src, int srcld, int N, int K,
                            half_t* __restrict__ dst, int Np, int Kp) {
  int idx = blockIdx.x * blockDim.x + threadIdx.x;
  if (idx >= Np * Kp) return;
  int rr = idx / Kp, cc = idx % Kp;
  float v = (rr < N && cc < K) ? src[(size_t)rr * srcld + cc] : 0.0f;
  dst[idx] = (half_t)v;
}

__global__ void k_sprof(const float* __restrict__ student_W, const int* __restrict__ s_id,
                        float* __restrict__ sprof) {
  int idx = blockIdx.x * blockDim.x + threadIdx.x;
  if (idx >= B_SZ * NP) return;
  int b = idx / NP, n = idx % NP;
  float v = 0.0f;
  if (n < NC) v = sigf(student_W[(size_t)s_id[b] * NC + n]);
  sprof[idx] = v;
}

// One block per (b,t) row; 256 threads cover the widest (EMB=256) segment.
__global__ void k_prep_rows(const float* __restrict__ q_maritx,
                            const float* __restrict__ q_maritx_next,
                            const int* __restrict__ e_id,
                            const int* __restrict__ r,
                            const int* __restrict__ q_next,
                            const float* __restrict__ k_diff_W,
                            const float* __restrict__ e_disc_W,
                            const float* __restrict__ answer_W,
                            const float* __restrict__ emb_problem,
                            const float* __restrict__ d_e,
                            half_t* __restrict__ q16,
                            float* __restrict__ kdiff,
                            float* __restrict__ edisc,
                            half_t* __restrict__ xh,
                            half_t* __restrict__ hcath) {
  size_t rw = blockIdx.x;
  int t = threadIdx.x;
  int e = e_id[rw];
  if (t < NP) {
    half_t qv = (half_t)0.0f;
    float kv = 0.0f;
    if (t < NC) {
      qv = (half_t)q_maritx[rw * NC + t];
      kv = sigf(k_diff_W[(size_t)e * NC + t]);
    }
    q16[rw * NP + t] = qv;
    kdiff[rw * NP + t] = kv;
  }
  if (t == 0) edisc[rw] = sigf(e_disc_W[e]) * d_e[0];
  int rr = r[rw];
  xh[rw * KP_X + NC + t] = (half_t)answer_W[(size_t)rr * EMBD + t];        // r_emb (shared x1/x2)
  if (t < KP_X - IN1) xh[rw * KP_X + IN1 + t] = (half_t)0.0f;              // K pad
  int qn = q_next[rw];
  hcath[rw * KP_H + NC + t] = (half_t)emb_problem[(size_t)qn * EMBD + t];  // e_next_emb
  if (t < NC) hcath[rw * KP_H + IN1 + t] = (half_t)q_maritx_next[rw * NC + t];
  if (t < KP_H - 502) hcath[rw * KP_H + 502 + t] = (half_t)0.0f;           // K pad
}

// ---------------- generic WMMA GEMM: D = epilogue(X * W^T + bias) -------------
// One wave computes a 16x128 tile (8 accumulators, A-fragment reused 8x).
enum GemmMode { MODE_F32 = 0, MODE_SIGF32 = 1, MODE_SIGF16 = 2, MODE_X1 = 3 };

template <int MODE>
__global__ void __launch_bounds__(32)
k_gemm(const half_t* __restrict__ X, const half_t* __restrict__ W,
       const float* __restrict__ bias, int Kp, int Nreal,
       void* __restrict__ outp, int ldo,
       const float* __restrict__ sprof) {
  const int lane = threadIdx.x;
  const int row0 = blockIdx.x * 16;
  const int col0 = blockIdx.y * 128;
  v8f acc[8];
  #pragma unroll
  for (int j = 0; j < 8; ++j) acc[j] = zero8();
  for (int kb = 0; kb < Kp; kb += 32) {
    if (kb + 32 < Kp)  // prefetch next K-block of A (global_prefetch_b8)
      __builtin_prefetch(X + (size_t)(row0 + (lane & 15)) * Kp + kb + 32, 0, 1);
    v16h a = load_frag(X, Kp, row0, kb, lane);
    #pragma unroll
    for (int j = 0; j < 8; ++j) {
      v16h b = load_frag(W, Kp, col0 + j * 16, kb, lane);
      acc[j] = __builtin_amdgcn_wmma_f32_16x16x32_f16(false, a, false, b,
                                                      (short)0, acc[j], false, false);
    }
  }
  const int nl = lane & 15;
  const int mhi = (lane >> 4) * 8;  // C/D: lanes 16..31 hold rows M+8
  #pragma unroll
  for (int j = 0; j < 8; ++j) {
    int n = col0 + j * 16 + nl;
    float bv = (n < Nreal) ? bias[n] : 0.0f;
    #pragma unroll
    for (int e = 0; e < 8; ++e) {
      size_t m = (size_t)row0 + e + mhi;
      float v = acc[j][e] + bv;
      if (MODE == MODE_F32) {
        ((float*)outp)[m * ldo + n] = v;
      } else if (MODE == MODE_SIGF32) {
        ((float*)outp)[m * ldo + n] = sigf(v);
      } else if (MODE == MODE_SIGF16) {
        ((half_t*)outp)[m * ldo + n] = (half_t)sigf(v);
      } else {  // MODE_X1: x1[:, :123] = (concept_ab) * s_prof  (f16)
        if (n < NC) {
          float sp = sprof[(m / T_SZ) * NP + n];
          ((half_t*)outp)[m * ldo + n] = (half_t)(v * sp);
        }
      }
    }
  }
}

// ---------------- scan 1: h = tanh(Xp + h W_hh^T + b_hh) * gate ----------------
// One wave per 16 batch rows; W_hh (128x128 f16) + h (16x128 f16) in LDS.
// Weights DMA'd in via the Tensor Data Mover when available.
__global__ void __launch_bounds__(32)
k_scan1(const half_t* __restrict__ whh16, const float* __restrict__ Xp,
        const float* __restrict__ b_hh, const float* __restrict__ W_tg,
        const float* __restrict__ b_tg, const float* __restrict__ sprof,
        const float* __restrict__ kdiff, const float* __restrict__ edisc,
        const float* __restrict__ q_maritx, const float* __restrict__ d_t,
        half_t* __restrict__ xh) {
  extern __shared__ char smem[];
  half_t* Wl = (half_t*)smem;                    // 128x128 f16 = 32KB @ LDS 0
  half_t* hl = (half_t*)(smem + NP * KP_C * 2);  // 16x128 f16 = 4KB @ LDS 32768
  int lane = threadIdx.x;
#ifdef USE_TDM
  tdm_load_lds(whh16, 0u, NP * KP_C * 2);        // TENSORcnt++
  {  // zero h while the DMA is in flight
    uint32_t* h = (uint32_t*)hl;
    for (int i = lane; i < 16 * KP_C / 2; i += 32) h[i] = 0u;
  }
  __builtin_amdgcn_s_wait_tensorcnt(0);
#else
  {
    const uint32_t* s = (const uint32_t*)whh16;
    uint32_t* d = (uint32_t*)Wl;
    for (int i = lane; i < NP * KP_C / 2; i += 32) d[i] = s[i];
    uint32_t* h = (uint32_t*)hl;
    for (int i = lane; i < 16 * KP_C / 2; i += 32) h[i] = 0u;
  }
#endif
  int b0 = blockIdx.x * 16;
  float dt = d_t[0];
  int nl = lane & 15;
  int koff = (lane >> 4) * 8;
  int mhi = (lane >> 4) * 8;
  for (int t = 0; t < T_SZ; ++t) {
    v8f acc[8];
    #pragma unroll
    for (int j = 0; j < 8; ++j) acc[j] = zero8();
    #pragma unroll
    for (int kb = 0; kb < KP_C; kb += 32) {
      F16Frag a;
      const half_t* ap = hl + nl * KP_C + kb + koff;
      a.h[0] = *(const h8*)ap;
      a.h[1] = *(const h8*)(ap + 16);
      #pragma unroll
      for (int j = 0; j < 8; ++j) {
        F16Frag b;
        const half_t* bp = Wl + (j * 16 + nl) * KP_C + kb + koff;
        b.h[0] = *(const h8*)bp;
        b.h[1] = *(const h8*)(bp + 16);
        acc[j] = __builtin_amdgcn_wmma_f32_16x16x32_f16(false, a.v, false, b.v,
                                                        (short)0, acc[j], false, false);
      }
    }
    #pragma unroll
    for (int j = 0; j < 8; ++j) {
      int n = j * 16 + nl;
      bool nv = (n < NC);
      float bh = nv ? b_hh[n] : 0.0f;
      float g = nv ? sigf((float)t * W_tg[n] + b_tg[n]) : 0.0f;
      #pragma unroll
      for (int e = 0; e < 8; ++e) {
        int m = e + mhi;
        int b = b0 + m;
        size_t row = (size_t)b * T_SZ + t;
        float hn = 0.0f;
        if (nv) {
          float pre = acc[j][e] + Xp[row * NP + n] + bh;
          hn = tanhfast(pre) * g;
          // fused epilogue: x2[:, :123] = e_disc*(d_t*s_prof+(1-d_t)*h - k_diff)*q
          float sp = sprof[(size_t)b * NP + n];
          float kd = kdiff[row * NP + n];
          float ed = edisc[row];
          float xv = ed * (dt * sp + (1.0f - dt) * hn - kd) * q_maritx[row * NC + n];
          xh[row * KP_X + n] = (half_t)xv;
        }
        hl[m * KP_C + n] = (half_t)hn;  // wave-private LDS: in-order, no barrier
      }
    }
  }
}

// ---------------- scan 2: GRU-style update ----------------
__global__ void __launch_bounds__(32)
k_scan2(const half_t* __restrict__ wzh16, const half_t* __restrict__ whh2_16,
        const float* __restrict__ Zx, const float* __restrict__ Hx,
        half_t* __restrict__ hcath) {
  extern __shared__ char smem[];
  half_t* Wz = (half_t*)smem;                                       // 32KB @ 0
  half_t* Wh = (half_t*)(smem + NP * KP_C * 2);                     // 32KB @ 32768
  float*  hf = (float*)(smem + 2 * NP * KP_C * 2);                  // 8KB  @ 65536
  half_t* hh = (half_t*)(smem + 2 * NP * KP_C * 2 + 16 * KP_C * 4); // 4KB  @ 73728
  int lane = threadIdx.x;
#ifdef USE_TDM
  tdm_load_lds(wzh16, 0u, NP * KP_C * 2);
  tdm_load_lds(whh2_16, (unsigned)(NP * KP_C * 2), NP * KP_C * 2);
  {
    for (int i = lane; i < 16 * KP_C; i += 32) hf[i] = 0.0f;
    uint32_t* h16 = (uint32_t*)hh;
    for (int i = lane; i < 16 * KP_C / 2; i += 32) h16[i] = 0u;
  }
  __builtin_amdgcn_s_wait_tensorcnt(0);
#else
  {
    const uint32_t* s1 = (const uint32_t*)wzh16;
    const uint32_t* s2 = (const uint32_t*)whh2_16;
    uint32_t* d1 = (uint32_t*)Wz;
    uint32_t* d2 = (uint32_t*)Wh;
    for (int i = lane; i < NP * KP_C / 2; i += 32) { d1[i] = s1[i]; d2[i] = s2[i]; }
    for (int i = lane; i < 16 * KP_C; i += 32) hf[i] = 0.0f;
    uint32_t* h16 = (uint32_t*)hh;
    for (int i = lane; i < 16 * KP_C / 2; i += 32) h16[i] = 0u;
  }
#endif
  int b0 = blockIdx.x * 16;
  int nl = lane & 15;
  int koff = (lane >> 4) * 8;
  int mhi = (lane >> 4) * 8;
  for (int t = 0; t < T_SZ; ++t) {
    v8f az[8], ah[8];
    #pragma unroll
    for (int j = 0; j < 8; ++j) { az[j] = zero8(); ah[j] = zero8(); }
    #pragma unroll
    for (int kb = 0; kb < KP_C; kb += 32) {
      F16Frag a;
      const half_t* ap = hh + nl * KP_C + kb + koff;
      a.h[0] = *(const h8*)ap;
      a.h[1] = *(const h8*)(ap + 16);
      #pragma unroll
      for (int j = 0; j < 8; ++j) {
        F16Frag bz, bh;
        const half_t* pz = Wz + (j * 16 + nl) * KP_C + kb + koff;
        const half_t* ph = Wh + (j * 16 + nl) * KP_C + kb + koff;
        bz.h[0] = *(const h8*)pz; bz.h[1] = *(const h8*)(pz + 16);
        bh.h[0] = *(const h8*)ph; bh.h[1] = *(const h8*)(ph + 16);
        az[j] = __builtin_amdgcn_wmma_f32_16x16x32_f16(false, a.v, false, bz.v,
                                                       (short)0, az[j], false, false);
        ah[j] = __builtin_amdgcn_wmma_f32_16x16x32_f16(false, a.v, false, bh.v,
                                                       (short)0, ah[j], false, false);
      }
    }
    #pragma unroll
    for (int j = 0; j < 8; ++j) {
      int n = j * 16 + nl;
      bool nv = (n < NC);
      #pragma unroll
      for (int e = 0; e < 8; ++e) {
        int m = e + mhi;
        size_t row = (size_t)(b0 + m) * T_SZ + t;
        float hn = 0.0f;
        if (nv) {
          float hp = hf[m * KP_C + n];
          float z = sigf(az[j][e] + Zx[row * NP + n]);      // b_z folded into Zx
          float cand = tanhfast(ah[j][e] + Hx[row * NP + n]);
          hn = (1.0f - z) * hp + z * cand;
          hcath[row * KP_H + n] = (half_t)hn;
        }
        hf[m * KP_C + n] = hn;
        hh[m * KP_C + n] = (half_t)hn;
      }
    }
  }
}

// ---------------- final 1-wide projection ----------------
__global__ void k_final(const float* __restrict__ o2, const float* __restrict__ p3_W,
                        const float* __restrict__ p3_b, float* __restrict__ out) {
  int row = blockIdx.x * blockDim.x + threadIdx.x;
  if (row >= BT) return;
  float acc = p3_b[0];
  const float* p = o2 + (size_t)row * NP;
  for (int n = 0; n < NC; ++n) acc += p[n] * p3_W[n];
  out[row] = sigf(acc);
}

// ---------------- host orchestration ----------------
extern "C" void kernel_launch(void* const* d_in, const int* in_sizes, int n_in,
                              void* d_out, int out_size, void* d_ws, size_t ws_size,
                              hipStream_t stream) {
  (void)in_sizes; (void)n_in; (void)out_size; (void)ws_size;
  const float* q_maritx      = (const float*)d_in[0];
  const float* q_maritx_next = (const float*)d_in[1];
  const int*   s_id          = (const int*)d_in[3];
  const int*   e_id          = (const int*)d_in[4];
  const int*   r             = (const int*)d_in[5];
  const int*   q_next        = (const int*)d_in[6];
  const float* emb_problem   = (const float*)d_in[8];
  const float* student_W     = (const float*)d_in[9];
  const float* k_diff_W      = (const float*)d_in[10];
  const float* e_disc_W      = (const float*)d_in[11];
  const float* answer_W      = (const float*)d_in[12];
  const float* zz_W          = (const float*)d_in[13];
  const float* zz_b          = (const float*)d_in[14];
  const float* d_t           = (const float*)d_in[15];
  const float* d_e           = (const float*)d_in[16];
  const float* W_ih          = (const float*)d_in[17];
  const float* b_ih          = (const float*)d_in[18];
  const float* W_hh          = (const float*)d_in[19];
  const float* b_hh          = (const float*)d_in[20];
  const float* W_tg          = (const float*)d_in[21];
  const float* b_tg          = (const float*)d_in[22];
  const float* W_z           = (const float*)d_in[23];
  const float* b_z           = (const float*)d_in[24];
  const float* W_h           = (const float*)d_in[25];
  const float* b_h           = (const float*)d_in[26];
  const float* p1_W          = (const float*)d_in[27];
  const float* p1_b          = (const float*)d_in[28];
  const float* p2_W          = (const float*)d_in[29];
  const float* p2_b          = (const float*)d_in[30];
  const float* p3_W          = (const float*)d_in[31];
  const float* p3_b          = (const float*)d_in[32];

  // workspace layout (bytes, all 256-aligned; total 144,568,320 ~ 138 MB)
  char* ws = (char*)d_ws;
  half_t* zz16  = (half_t*)(ws + 0);          // 128x128
  half_t* wih16 = (half_t*)(ws + 32768);      // 128x384
  half_t* wzx16 = (half_t*)(ws + 131072);     // 128x384
  half_t* whx16 = (half_t*)(ws + 229376);     // 128x384
  half_t* whh16 = (half_t*)(ws + 327680);     // 128x128
  half_t* wzh16 = (half_t*)(ws + 360448);     // 128x128
  half_t* whh2  = (half_t*)(ws + 393216);     // 128x128
  half_t* p116  = (half_t*)(ws + 425984);     // 256x512
  half_t* p216  = (half_t*)(ws + 688128);     // 128x256
  float*  sprof = (float*)(ws + 753664);      // B x 128
  float*  edisc = (float*)(ws + 1277952);     // BT
  float*  kdiff = (float*)(ws + 1437696);     // BT x 128 f32
  half_t* q16   = (half_t*)(ws + 21884928);   // BT x 128 f16
  half_t* xh    = (half_t*)(ws + 32108544);   // BT x 384 f16 (x1, then x2 in place)
  float*  Xp    = (float*)(ws + 62779392);    // BT x 128 f32
  half_t* hcath = (half_t*)(ws + 83226624);   // BT x 512 f16
  half_t* o1h   = (half_t*)(ws + 124121088);  // BT x 256 f16
  float*  Zx    = Xp;            // alias: Xp dead after scan1
  float*  Hx    = kdiff;         // alias: kdiff dead after scan1
  float*  o2    = (float*)xh;    // alias: xh dead after Zx/Hx GEMMs

  dim3 b32(32);
  auto cvt = [&](const float* src, int srcld, int N, int K, half_t* dst, int Np, int Kp) {
    int total = Np * Kp;
    k_convert_w<<<(total + 255) / 256, 256, 0, stream>>>(src, srcld, N, K, dst, Np, Kp);
  };
  cvt(zz_W, NC, NC, NC, zz16, NP, KP_C);
  cvt(W_ih, IN1, NC, IN1, wih16, NP, KP_X);
  cvt(W_z, 502, NC, IN1, wzx16, NP, KP_X);        // W_z[:, :379]
  cvt(W_h, 502, NC, IN1, whx16, NP, KP_X);        // W_h[:, :379]
  cvt(W_hh, NC, NC, NC, whh16, NP, KP_C);
  cvt(W_z + IN1, 502, NC, NC, wzh16, NP, KP_C);   // W_z[:, 379:]
  cvt(W_h + IN1, 502, NC, NC, whh2, NP, KP_C);    // W_h[:, 379:]
  cvt(p1_W, 502, 256, 502, p116, 256, KP_H);
  cvt(p2_W, 256, NC, 256, p216, NP, 256);

  k_sprof<<<(B_SZ * NP + 255) / 256, 256, 0, stream>>>(student_W, s_id, sprof);
  k_prep_rows<<<BT, 256, 0, stream>>>(q_maritx, q_maritx_next, e_id, r, q_next,
                                      k_diff_W, e_disc_W, answer_W, emb_problem, d_e,
                                      q16, kdiff, edisc, xh, hcath);

  dim3 g128(BT / 16, 1);   // N=128 (one 16x128 tile per wave)
  dim3 g256(BT / 16, 2);   // N=256
  // concept_ab = q*zz^T + zz_b, scaled by s_prof -> x1 cols [0,123)
  k_gemm<MODE_X1><<<g128, b32, 0, stream>>>(q16, zz16, zz_b, KP_C, NC, xh, KP_X, sprof);
  // Xp = x1 * W_ih^T + b_ih
  k_gemm<MODE_F32><<<g128, b32, 0, stream>>>(xh, wih16, b_ih, KP_X, NC, Xp, NP, nullptr);
  // recurrence 1 (also writes x2 cols [0,123) in place)
  k_scan1<<<B_SZ / 16, b32, (NP * KP_C + 16 * KP_C) * 2, stream>>>(
      whh16, Xp, b_hh, W_tg, b_tg, sprof, kdiff, edisc, q_maritx, d_t, xh);
  // Zx / Hx = x2 * W{z,h}_x^T + b
  k_gemm<MODE_F32><<<g128, b32, 0, stream>>>(xh, wzx16, b_z, KP_X, NC, Zx, NP, nullptr);
  k_gemm<MODE_F32><<<g128, b32, 0, stream>>>(xh, whx16, b_h, KP_X, NC, Hx, NP, nullptr);
  // recurrence 2 (writes hcat cols [0,123))
  k_scan2<<<B_SZ / 16, b32, 2 * NP * KP_C * 2 + 16 * KP_C * 4 + 16 * KP_C * 2, stream>>>(
      wzh16, whh2, Zx, Hx, hcath);
  // o1 = sigmoid(hcat * p1^T + p1_b) -> f16
  k_gemm<MODE_SIGF16><<<g256, b32, 0, stream>>>(hcath, p116, p1_b, KP_H, 256, o1h, 256, nullptr);
  // o2 = sigmoid(o1 * p2^T + p2_b) -> f32
  k_gemm<MODE_SIGF32><<<g128, b32, 0, stream>>>(o1h, p216, p2_b, 256, NC, o2, NP, nullptr);
  // out = sigmoid(o2 * p3^T + p3_b)
  k_final<<<(BT + 255) / 256, 256, 0, stream>>>(o2, p3_W, p3_b, (float*)d_out);
}
